// MultiHeadAttention_31241592111289
// MI455X (gfx1250) — matrix operand order repaired and checked by reading, not compiled, not run on hardware
//
#include <hip/hip_runtime.h>

// ---------------- types for WMMA fragments (gfx1250, wave32) ----------------
typedef __attribute__((ext_vector_type(16))) __bf16 v16bf;
typedef __attribute__((ext_vector_type(8)))  float  v8f;

union FragAB {
    v16bf v;
    uint4 q[2];
    unsigned short h[16];
};

__device__ __forceinline__ unsigned short f2bf(float f) {
    unsigned u = __float_as_uint(f);
    unsigned r = u + 0x7FFFu + ((u >> 16) & 1u);   // round-to-nearest-even
    return (unsigned short)(r >> 16);
}

__device__ __forceinline__ void cvt8(const float4 a, const float4 b, unsigned short* h) {
    h[0] = f2bf(a.x); h[1] = f2bf(a.y); h[2] = f2bf(a.z); h[3] = f2bf(a.w);
    h[4] = f2bf(b.x); h[5] = f2bf(b.y); h[6] = f2bf(b.z); h[7] = f2bf(b.w);
}

#define WMMA_BF16(A, B, C) \
    __builtin_amdgcn_wmma_f32_16x16x32_bf16(false, (A), false, (B), (short)0, (C), false, false)

// ---------------- elementwise f32 -> bf16 (4-wide) ----------------
__global__ void __launch_bounds__(256) mha_cvt_bf16x4(const float4* __restrict__ src,
                                                      uint2* __restrict__ dst, int n4) {
    int i = blockIdx.x * blockDim.x + threadIdx.x;
    if (i < n4) {
        float4 v = src[i];
        uint2 o;
        o.x = (unsigned)f2bf(v.x) | ((unsigned)f2bf(v.y) << 16);
        o.y = (unsigned)f2bf(v.z) | ((unsigned)f2bf(v.w) << 16);
        dst[i] = o;
    }
}

// ---------------- weight transpose+convert: W[1024][1024] f32 -> Wt[n][k] bf16 ----------------
__global__ void __launch_bounds__(256) mha_wtrans(const float* __restrict__ W,
                                                  unsigned short* __restrict__ Wt) {
    __shared__ float tile[32][33];
    const int k0 = blockIdx.y * 32, n0 = blockIdx.x * 32;
    const int tx = threadIdx.x & 31, ty = threadIdx.x >> 5;   // 32 x 8
    #pragma unroll
    for (int j = 0; j < 32; j += 8)
        tile[ty + j][tx] = W[(size_t)(k0 + ty + j) * 1024 + (n0 + tx)];
    __syncthreads();
    #pragma unroll
    for (int j = 0; j < 32; j += 8)
        Wt[(size_t)(n0 + ty + j) * 1024 + (k0 + tx)] = f2bf(tile[tx][ty + j]);
}

// ---------------- generic bf16 GEMM: 16x64 per wave, double-buffered K-loop by 32 ----------------
// A: [M,Kc] bf16 row-major.  Bt: [N,Kc] bf16 (each output column contiguous along K).
// out_mode 0: head-major bf16  ((b*16+h)*2048+s)*64+d      (Q,K projections)
// out_mode 1: head-transposed bf16 ((b*16+h)*64+d)*2048+s  (V projection)
// out_mode 2: plain f32 row-major [M,Ncols]                (final output)
__global__ void __launch_bounds__(128) mha_gemm_bf16(const unsigned short* __restrict__ A,
                                                     const unsigned short* __restrict__ Bt,
                                                     int Kc, int Ncols, int out_mode,
                                                     void* __restrict__ out, float scale) {
    const int lane  = threadIdx.x & 31;
    const int wave  = threadIdx.x >> 5;
    const int mtile = blockIdx.y;
    const int ncol0 = blockIdx.x * 256 + wave * 64;   // wave covers 64 output cols
    const int m     = lane & 15;
    const int khalf = lane >> 4;

    const unsigned short* arow = A  + (size_t)(mtile * 16 + m) * Kc;
    const unsigned short* bcol = Bt + (size_t)(ncol0 + m) * Kc;   // tile t at +t*16*Kc

    v8f acc[4] = {};
    FragAB a0, b0[4];

    // prologue: fetch K-step 0
    a0.q[0] = *(const uint4*)(arow + 16 * khalf);
    a0.q[1] = *(const uint4*)(arow + 8 + 16 * khalf);
    #pragma unroll
    for (int t = 0; t < 4; ++t) {
        const uint4* bp = (const uint4*)(bcol + (size_t)t * 16 * Kc + 16 * khalf);
        b0[t].q[0] = bp[0];
        b0[t].q[1] = bp[1];
    }

    for (int c0 = 32; c0 < Kc; c0 += 32) {
        __builtin_prefetch(arow + c0 + 128, 0, 0);
        // fetch next K-step while current one computes
        FragAB a1, b1[4];
        a1.q[0] = *(const uint4*)(arow + c0 + 16 * khalf);
        a1.q[1] = *(const uint4*)(arow + c0 + 8 + 16 * khalf);
        #pragma unroll
        for (int t = 0; t < 4; ++t) {
            const uint4* bp = (const uint4*)(bcol + (size_t)t * 16 * Kc + c0 + 16 * khalf);
            b1[t].q[0] = bp[0];
            b1[t].q[1] = bp[1];
        }
        #pragma unroll
        for (int t = 0; t < 4; ++t) acc[t] = WMMA_BF16(a0.v, b0[t].v, acc[t]);
        a0 = a1;
        #pragma unroll
        for (int t = 0; t < 4; ++t) b0[t] = b1[t];
    }
    #pragma unroll
    for (int t = 0; t < 4; ++t) acc[t] = WMMA_BF16(a0.v, b0[t].v, acc[t]);

    #pragma unroll
    for (int t = 0; t < 4; ++t) {
        const int gcol = ncol0 + t * 16 + m;
        if (out_mode == 2) {
            float* O = (float*)out;
            #pragma unroll
            for (int r = 0; r < 8; ++r) {
                int grow = mtile * 16 + r + 8 * khalf;
                O[(size_t)grow * Ncols + gcol] = acc[t][r] * scale;
            }
        } else {
            unsigned short* O = (unsigned short*)out;
            const int h = gcol >> 6, d = gcol & 63;
            #pragma unroll
            for (int r = 0; r < 8; ++r) {
                int grow = mtile * 16 + r + 8 * khalf;
                int bb = grow >> 11, s = grow & 2047;
                size_t idx = (out_mode == 0)
                    ? ((size_t)(bb * 16 + h) * 2048 + s) * 64 + d
                    : ((size_t)(bb * 16 + h) * 64 + d) * 2048 + s;
                O[idx] = f2bf(acc[t][r] * scale);
            }
        }
    }
}

// ---------------- scores = Qh . Kh^T : 16x64 per wave, Q frag reused across 4 K-tiles ----------------
__global__ void __launch_bounds__(128) mha_scores(const unsigned short* __restrict__ Qh,
                                                  const unsigned short* __restrict__ Kh,
                                                  const unsigned char* __restrict__ mask,
                                                  float* __restrict__ attn) {
    const int lane  = threadIdx.x & 31;
    const int wave  = threadIdx.x >> 5;
    const int bh    = blockIdx.z;
    const int qtile = blockIdx.y;
    const int kbase = (blockIdx.x * 4 + wave) * 64;   // wave covers 64 key cols
    const int m     = lane & 15;
    const int khalf = lane >> 4;

    const unsigned short* qrow = Qh + ((size_t)bh * 2048 + qtile * 16 + m) * 64;
    const unsigned short* krow = Kh + ((size_t)bh * 2048 + kbase + m) * 64;  // tile t at +t*16*64

    v8f acc[4] = {};
    #pragma unroll
    for (int c0 = 0; c0 < 64; c0 += 32) {
        FragAB a;
        a.q[0] = *(const uint4*)(qrow + c0 + 16 * khalf);
        a.q[1] = *(const uint4*)(qrow + c0 + 8 + 16 * khalf);
        #pragma unroll
        for (int t = 0; t < 4; ++t) {
            FragAB b;
            const uint4* bp = (const uint4*)(krow + (size_t)t * 16 * 64 + c0 + 16 * khalf);
            b.q[0] = bp[0];
            b.q[1] = bp[1];
            acc[t] = WMMA_BF16(a.v, b.v, acc[t]);
        }
    }

    const int b = bh >> 4;
    #pragma unroll
    for (int t = 0; t < 4; ++t) {
        const int kcol = kbase + t * 16 + m;
        #pragma unroll
        for (int r = 0; r < 8; ++r) {
            int q = qtile * 16 + r + 8 * khalf;
            float v = acc[t][r];
            if (mask[((size_t)b * 2048 + q) * 2048 + kcol]) v = -1e9f;
            attn[((size_t)bh * 2048 + q) * 2048 + kcol] = v;
        }
    }
}

// ---------------- in-place softmax over rows of length 2048 ----------------
__global__ void __launch_bounds__(256) mha_softmax(float* __restrict__ attn) {
    __shared__ float red[256];
    float* p = attn + (size_t)blockIdx.x * 2048;
    const int t = threadIdx.x;
    float x[8];
    float mx = -3.4e38f;
    #pragma unroll
    for (int j = 0; j < 8; ++j) { x[j] = p[t + j * 256]; mx = fmaxf(mx, x[j]); }
    red[t] = mx; __syncthreads();
    for (int off = 128; off > 0; off >>= 1) {
        if (t < off) red[t] = fmaxf(red[t], red[t + off]);
        __syncthreads();
    }
    mx = red[0]; __syncthreads();
    float s = 0.f;
    #pragma unroll
    for (int j = 0; j < 8; ++j) { x[j] = __expf(x[j] - mx); s += x[j]; }
    red[t] = s; __syncthreads();
    for (int off = 128; off > 0; off >>= 1) {
        if (t < off) red[t] += red[t + off];
        __syncthreads();
    }
    const float inv = 1.0f / red[0];
    #pragma unroll
    for (int j = 0; j < 8; ++j) p[t + j * 256] = x[j] * inv;
}

// ---------------- context = attn . V : 16x64 per wave, double-buffered, cvt amortized 4x ----------------
__global__ void __launch_bounds__(128) mha_context(const float* __restrict__ attn,
                                                   const unsigned short* __restrict__ Vt,
                                                   unsigned short* __restrict__ Ctx) {
    const int lane  = threadIdx.x & 31;
    const int wave  = threadIdx.x >> 5;
    const int bh    = blockIdx.z;
    const int qtile = blockIdx.y * 4 + wave;   // each wave owns one q-tile, all 64 d cols
    const int m     = lane & 15;
    const int khalf = lane >> 4;

    const float*          arow = attn + ((size_t)bh * 2048 + qtile * 16 + m) * 2048;
    const unsigned short* bcol = Vt   + ((size_t)bh * 64 + m) * 2048;  // tile t at +t*16*2048

    v8f acc[4] = {};
    float4 fa[4];
    FragAB b0[4];

    // prologue: fetch K-step 0
    {
        const float4* f0 = (const float4*)(arow + 16 * khalf);
        const float4* f1 = (const float4*)(arow + 8 + 16 * khalf);
        fa[0] = f0[0]; fa[1] = f0[1]; fa[2] = f1[0]; fa[3] = f1[1];
        #pragma unroll
        for (int t = 0; t < 4; ++t) {
            const uint4* bp = (const uint4*)(bcol + (size_t)t * 16 * 2048 + 16 * khalf);
            b0[t].q[0] = bp[0];
            b0[t].q[1] = bp[1];
        }
    }

    for (int c0 = 32; c0 < 2048; c0 += 32) {
        __builtin_prefetch(arow + c0 + 128, 0, 0);
        // fetch next K-step
        float4 fn[4];
        FragAB b1[4];
        const float4* f0 = (const float4*)(arow + c0 + 16 * khalf);
        const float4* f1 = (const float4*)(arow + c0 + 8 + 16 * khalf);
        fn[0] = f0[0]; fn[1] = f0[1]; fn[2] = f1[0]; fn[3] = f1[1];
        #pragma unroll
        for (int t = 0; t < 4; ++t) {
            const uint4* bp = (const uint4*)(bcol + (size_t)t * 16 * 2048 + c0 + 16 * khalf);
            b1[t].q[0] = bp[0];
            b1[t].q[1] = bp[1];
        }
        // compute current K-step
        FragAB a;
        cvt8(fa[0], fa[1], &a.h[0]);
        cvt8(fa[2], fa[3], &a.h[8]);
        #pragma unroll
        for (int t = 0; t < 4; ++t) acc[t] = WMMA_BF16(a.v, b0[t].v, acc[t]);
        fa[0] = fn[0]; fa[1] = fn[1]; fa[2] = fn[2]; fa[3] = fn[3];
        #pragma unroll
        for (int t = 0; t < 4; ++t) b0[t] = b1[t];
    }
    {
        FragAB a;
        cvt8(fa[0], fa[1], &a.h[0]);
        cvt8(fa[2], fa[3], &a.h[8]);
        #pragma unroll
        for (int t = 0; t < 4; ++t) acc[t] = WMMA_BF16(a.v, b0[t].v, acc[t]);
    }

    const int b = bh >> 4, h = bh & 15;
    #pragma unroll
    for (int t = 0; t < 4; ++t) {
        const int col = h * 64 + t * 16 + m;
        #pragma unroll
        for (int r = 0; r < 8; ++r) {
            int q = qtile * 16 + r + 8 * khalf;
            Ctx[((size_t)(b * 2048 + q)) * 1024 + col] = f2bf(acc[t][r]);
        }
    }
}

// ---------------- host-side orchestration ----------------
extern "C" void kernel_launch(void* const* d_in, const int* in_sizes, int n_in,
                              void* d_out, int out_size, void* d_ws, size_t ws_size,
                              hipStream_t stream) {
    const float* inQ  = (const float*)d_in[0];
    const float* inK  = (const float*)d_in[1];
    const float* inV  = (const float*)d_in[2];
    const unsigned char* mask = (const unsigned char*)d_in[3];
    const float* WQ   = (const float*)d_in[4];
    const float* WK   = (const float*)d_in[5];
    const float* WV   = (const float*)d_in[6];
    const float* WFC  = (const float*)d_in[7];

    char* ws = (char*)d_ws;
    const size_t MB = 1024 * 1024;
    unsigned short* XbQ = (unsigned short*)(ws + 0 * MB);   // 8 MB
    unsigned short* XbK = (unsigned short*)(ws + 8 * MB);   // 8 MB
    unsigned short* XbV = (unsigned short*)(ws + 16 * MB);  // 8 MB
    unsigned short* WtQ = (unsigned short*)(ws + 24 * MB);  // 2 MB each
    unsigned short* WtK = (unsigned short*)(ws + 26 * MB);
    unsigned short* WtV = (unsigned short*)(ws + 28 * MB);
    unsigned short* WtF = (unsigned short*)(ws + 30 * MB);
    unsigned short* Qh  = (unsigned short*)(ws + 32 * MB);  // [B*H][2048][64] bf16
    unsigned short* Kh  = (unsigned short*)(ws + 40 * MB);  // [B*H][2048][64] bf16
    unsigned short* Vt  = (unsigned short*)(ws + 48 * MB);  // [B*H][64][2048] bf16
    unsigned short* Ctx = XbQ;                               // reuse: free after Q proj

    float* outO = (float*)d_out;                             // [2,2048,1024]
    float* attn = outO + (size_t)4096 * 1024;                // [2,16,2048,2048]

    // 1. inputs f32 -> bf16
    const int n4 = (4096 * 1024) / 4;
    dim3 cb(256), cg((n4 + 255) / 256);
    mha_cvt_bf16x4<<<cg, cb, 0, stream>>>((const float4*)inQ, (uint2*)XbQ, n4);
    mha_cvt_bf16x4<<<cg, cb, 0, stream>>>((const float4*)inK, (uint2*)XbK, n4);
    mha_cvt_bf16x4<<<cg, cb, 0, stream>>>((const float4*)inV, (uint2*)XbV, n4);

    // 2. weights -> transposed bf16 [n][k]
    dim3 tg(32, 32), tb(256);
    mha_wtrans<<<tg, tb, 0, stream>>>(WQ,  WtQ);
    mha_wtrans<<<tg, tb, 0, stream>>>(WK,  WtK);
    mha_wtrans<<<tg, tb, 0, stream>>>(WV,  WtV);
    mha_wtrans<<<tg, tb, 0, stream>>>(WFC, WtF);

    // 3. projections (scale 1/sqrt(64) folded into Q)
    dim3 gb(128), gg(1024 / 256, 4096 / 16);
    mha_gemm_bf16<<<gg, gb, 0, stream>>>(XbQ, WtQ, 1024, 1024, 0, Qh, 0.125f);
    mha_gemm_bf16<<<gg, gb, 0, stream>>>(XbK, WtK, 1024, 1024, 0, Kh, 1.0f);
    mha_gemm_bf16<<<gg, gb, 0, stream>>>(XbV, WtV, 1024, 1024, 1, Vt, 1.0f);

    // 4. raw scores (+mask) straight into the attn output region
    dim3 sg(8, 128, 32), sb(128);
    mha_scores<<<sg, sb, 0, stream>>>(Qh, Kh, mask, attn);

    // 5. softmax in place over 2048-long rows
    mha_softmax<<<dim3(65536), dim3(256), 0, stream>>>(attn);

    // 6. context = attn . V  -> bf16 [B,S,H*dv]
    dim3 xg(1, 32, 32), xb(128);
    mha_context<<<xg, xb, 0, stream>>>(attn, Vt, Ctx);

    // 7. output = context . W_fc  -> f32 [B,S,D]
    mha_gemm_bf16<<<gg, gb, 0, stream>>>(Ctx, WtF, 1024, 1024, 2, outO, 1.0f);
}
